// DeepseekV2DecoderLayer_53609781788759
// MI455X (gfx1250) — matrix operand order repaired and checked by reading, not compiled, not run on hardware
//
#include <hip/hip_runtime.h>
#include <hip/hip_bf16.h>
#include <cstdint>

typedef __bf16 bf16;
typedef __attribute__((ext_vector_type(16))) __bf16 v16bf;
typedef __attribute__((ext_vector_type(8)))  __bf16 v8bf;
typedef __attribute__((ext_vector_type(4)))  __bf16 v4bf;
typedef __attribute__((ext_vector_type(8)))  float  v8f;

#if defined(__HIP_DEVICE_COMPILE__) && __has_builtin(__builtin_amdgcn_tensor_load_to_lds)
#define HAVE_TDM 1
#else
#define HAVE_TDM 0
#endif

// Problem constants (match reference)
constexpr int TOK = 2048;
constexpr int HID = 5120;
constexpr int NH  = 32;
constexpr int DN  = 128;
constexpr int DR  = 64;
constexpr int DV  = 128;
constexpr int QL  = 1536;
constexpr int KL  = 512;
constexpr int QKV_N = QL + KL + DR;   // 2112
constexpr int QD  = DN + DR;          // 192
constexpr int KVD = DN + DV;          // 256

__device__ __forceinline__ v8f v8f_zero() {
    v8f z;
#pragma unroll
    for (int i = 0; i < 8; ++i) z[i] = 0.0f;
    return z;
}

__device__ __forceinline__ v4bf cvt4(float4 f) {
    v4bf b;
    b[0] = (bf16)f.x; b[1] = (bf16)f.y; b[2] = (bf16)f.z; b[3] = (bf16)f.w;
    return b;
}

// A fragment (16-bit A 16x32): elems 0..7 <-> K=kha..kha+7, 8..15 <-> +16.
__device__ __forceinline__ v16bf load_frag_a(const bf16* base) {
    union { v16bf v; v8bf h[2]; } u;
    u.h[0] = *(const v8bf*)(base);
    u.h[1] = *(const v8bf*)(base + 16);
    return u.v;
}
// B fragment (16-bit B 32x16): elems 0..7 <-> K=kha..kha+7, 8..15 <-> +16.
__device__ __forceinline__ v16bf load_frag_b(const bf16* base) {
    union { v16bf v; v8bf h[2]; } u;
    u.h[0] = *(const v8bf*)(base);
    u.h[1] = *(const v8bf*)(base + 16);
    return u.v;
}

#if HAVE_TDM
typedef unsigned int u32x4 __attribute__((ext_vector_type(4)));
typedef int          i32x8 __attribute__((ext_vector_type(8)));
typedef int          i32x4 __attribute__((ext_vector_type(4)));

// Issue one 2-D bf16 tile load via the Tensor Data Mover.
// tile: tile_d0 elems (contig, dim0) x tile_d1 rows; row stride = stride0
// elems; LDS rows padded: pad_interval/pad_amount are D# encodings.
__device__ __forceinline__ void tdm_load_2d(
    unsigned lds_addr, const void* gaddr,
    unsigned tdim0, unsigned tdim1, unsigned stride0,
    unsigned tile_d0, unsigned tile_d1,
    unsigned pad_interval, unsigned pad_amount)
{
    unsigned long long ga = (unsigned long long)(uintptr_t)gaddr;
    u32x4 g0;
    g0[0] = 1u;                                   // count=1 (valid descriptor)
    g0[1] = lds_addr;                             // LDS byte address
    g0[2] = (unsigned)ga;                         // global_addr[31:0]
    g0[3] = (unsigned)((ga >> 32) & 0x01FFFFFFu) | 0x80000000u;  // addr hi | type=2
    i32x8 g1;
    g1[0] = (int)((1u << 16) | (1u << 20) |       // data_size=2B, pad_enable
                  (pad_interval << 22) | (pad_amount << 25));
    g1[1] = (int)((tdim0 & 0xFFFFu) << 16);                        // dim0[15:0]
    g1[2] = (int)((tdim0 >> 16) | ((tdim1 & 0xFFFFu) << 16));      // dim0 hi|dim1 lo
    g1[3] = (int)((tdim1 >> 16) | (tile_d0 << 16));                // dim1 hi|tile0
    g1[4] = (int)tile_d1;                                          // tile1, tile2=0
    g1[5] = (int)stride0;                                          // stride0[31:0]
    g1[6] = 0;
    g1[7] = 0;
    i32x4 zz4; zz4[0] = 0; zz4[1] = 0; zz4[2] = 0; zz4[3] = 0;
    i32x8 zz8;
#pragma unroll
    for (int i = 0; i < 8; ++i) zz8[i] = 0;
    __builtin_amdgcn_tensor_load_to_lds(g0, g1, zz4, zz4, zz8, 0);
}
#endif

// ---------------------------------------------------------------------------
// bf16 WMMA GEMM: C[M,N] = A[M,K] @ Bt[N,K]^T  (Bt is pre-transposed weights)
// Block tile 128x128, K step 32, double-buffered LDS; both operands staged as
// raw 2-D bf16 tiles -> TDM when available, v8bf copies otherwise.
// Requires: M % 128 == 0, K % 32 == 0.
// ---------------------------------------------------------------------------
template <typename OT>
__global__ __launch_bounds__(256) void gemm_bf16(
    const bf16* __restrict__ A, const bf16* __restrict__ Bt,
    OT* __restrict__ C, int M, int N, int K)
{
    __shared__ __align__(16) bf16 As[2][128][48];   // [row][k], 96B rows
    __shared__ __align__(16) bf16 Bs[2][128][48];   // [col][k], 96B rows

    const int tid  = threadIdx.x;
    const int lane = tid & 31;
    const int wid  = tid >> 5;
    const int wm   = wid & 3;
    const int wn   = wid >> 2;
    const int l16  = lane & 15;
    const int kha  = (lane < 16) ? 0 : 8;
    const int khb  = (lane < 16) ? 0 : 16;
    const int block_row = blockIdx.y * 128;
    const int block_col = blockIdx.x * 128;

    auto stage = [&](int buf, int k0) {
#if HAVE_TDM
        if (wid == 0) {
            // 32 elems (64B=16DW) per row -> pad 8DW(32B) => 96B LDS rows
            tdm_load_2d((unsigned)(uintptr_t)&As[buf][0][0],
                        A + (size_t)block_row * K + k0,
                        (unsigned)(K - k0), (unsigned)(M - block_row),
                        (unsigned)K, 32u, 128u, 3u, 7u);
            tdm_load_2d((unsigned)(uintptr_t)&Bs[buf][0][0],
                        Bt + (size_t)block_col * K + k0,
                        (unsigned)(K - k0), (unsigned)(N - block_col),
                        (unsigned)K, 32u, 128u, 3u, 7u);
        }
#else
        for (int idx = tid; idx < 128 * 4; idx += 256) {
            int r = idx >> 2, c8 = (idx & 3) * 8;
            *(v8bf*)&As[buf][r][c8] =
                *(const v8bf*)(A + (size_t)(block_row + r) * K + k0 + c8);
            int br = block_col + r;
            v8bf bv;
            if (br < N) bv = *(const v8bf*)(Bt + (size_t)br * K + k0 + c8);
            else { for (int e = 0; e < 8; ++e) bv[e] = (bf16)0.0f; }
            *(v8bf*)&Bs[buf][r][c8] = bv;
        }
#endif
    };
    auto stage_wait = [&]() {
#if HAVE_TDM
        if (wid == 0) __builtin_amdgcn_s_wait_tensorcnt(0);
#endif
        __syncthreads();
    };

    v8f acc[2][4];
#pragma unroll
    for (int mt = 0; mt < 2; ++mt)
#pragma unroll
        for (int nt = 0; nt < 4; ++nt) acc[mt][nt] = v8f_zero();

    const int nk = K >> 5;
    stage(0, 0);
    stage_wait();

    for (int kt = 0; kt < nk; ++kt) {
        const int buf = kt & 1;
        const bool pre = (kt + 1 < nk);
        if (pre) stage(buf ^ 1, (kt + 1) * 32);

        v16bf afr[2], bfr[4];
#pragma unroll
        for (int mt = 0; mt < 2; ++mt)
            afr[mt] = load_frag_a(&As[buf][wm * 32 + mt * 16 + l16][kha]);
#pragma unroll
        for (int nt = 0; nt < 4; ++nt)
            bfr[nt] = load_frag_b(&Bs[buf][wn * 64 + nt * 16 + l16][kha]);
#pragma unroll
        for (int mt = 0; mt < 2; ++mt)
#pragma unroll
            for (int nt = 0; nt < 4; ++nt)
                acc[mt][nt] = __builtin_amdgcn_wmma_f32_16x16x32_bf16(
                    false, afr[mt], false, bfr[nt], (short)0, acc[mt][nt],
                    false, false);
        if (pre) stage_wait();
    }

    const int rh = (lane < 16) ? 0 : 8;
#pragma unroll
    for (int mt = 0; mt < 2; ++mt)
#pragma unroll
        for (int nt = 0; nt < 4; ++nt) {
            int col = block_col + wn * 64 + nt * 16 + l16;
            if (col < N) {
#pragma unroll
                for (int i = 0; i < 8; ++i) {
                    int row = block_row + wm * 32 + mt * 16 + rh + i;
                    C[(size_t)row * N + col] = (OT)acc[mt][nt][i];
                }
            }
        }
}

// ---------------------------------------------------------------------------
// Weight prep: transpose fp32 [K,N] -> bf16 [N,K]
// ---------------------------------------------------------------------------
__global__ __launch_bounds__(256) void transpose_cvt(
    const float* __restrict__ W, bf16* __restrict__ Wt, int K, int N)
{
    __shared__ float tile[32][33];
    const int kb = blockIdx.y * 32, nb = blockIdx.x * 32;
    const int r = threadIdx.x >> 5, c = threadIdx.x & 31;
    for (int rr = r; rr < 32; rr += 8) {
        int k = kb + rr, n = nb + c;
        tile[rr][c] = (k < K && n < N) ? W[(size_t)k * N + n] : 0.0f;
    }
    __syncthreads();
    for (int rr = r; rr < 32; rr += 8) {
        int n = nb + rr, k = kb + c;
        if (n < N && k < K) Wt[(size_t)n * K + k] = (bf16)tile[c][rr];
    }
}

// Elementwise fp32 -> bf16 (n % 4 == 0)
__global__ __launch_bounds__(256) void cvt_f32_bf16(
    const float* __restrict__ x, bf16* __restrict__ y, int n4)
{
    for (int i = blockIdx.x * 256 + threadIdx.x; i < n4; i += gridDim.x * 256) {
        float4 f = *(const float4*)(x + (size_t)i * 4);
        *(v4bf*)(y + (size_t)i * 4) = cvt4(f);
    }
}

// ---------------------------------------------------------------------------
// Fused RMSNorm(q_a)->bf16, RMSNorm(kv_a)->bf16, RoPE(k_pe)->bf16.
// ---------------------------------------------------------------------------
__global__ __launch_bounds__(256) void prep_norm_rope(
    const float* __restrict__ qkv, const float* __restrict__ q_ln,
    const float* __restrict__ kv_ln, const int* __restrict__ positions,
    bf16* __restrict__ qan, bf16* __restrict__ kvan, bf16* __restrict__ kpe)
{
    __shared__ float red[256];
    const int t   = blockIdx.x;
    const int tid = threadIdx.x;
    const float* row = qkv + (size_t)t * QKV_N;

    float ss = 0.0f;
    for (int c = tid; c < QL; c += 256) { float v = row[c]; ss += v * v; }
    red[tid] = ss; __syncthreads();
    for (int s = 128; s > 0; s >>= 1) {
        if (tid < s) red[tid] += red[tid + s];
        __syncthreads();
    }
    float rms_q = rsqrtf(red[0] / (float)QL + 1e-6f);
    __syncthreads();

    ss = 0.0f;
    for (int c = tid; c < KL; c += 256) { float v = row[QL + c]; ss += v * v; }
    red[tid] = ss; __syncthreads();
    for (int s = 128; s > 0; s >>= 1) {
        if (tid < s) red[tid] += red[tid + s];
        __syncthreads();
    }
    float rms_kv = rsqrtf(red[0] / (float)KL + 1e-6f);
    __syncthreads();

    for (int c = tid; c < QL; c += 256)
        qan[(size_t)t * QL + c] = (bf16)(row[c] * rms_q * q_ln[c]);
    for (int c = tid; c < KL; c += 256)
        kvan[(size_t)t * KL + c] = (bf16)(row[QL + c] * rms_kv * kv_ln[c]);

    if (tid < DR / 2) {
        float pos = (float)positions[t];
        int i = tid;
        float inv = __expf(-((2.0f * i) / (float)DR) * 9.210340371976184f);
        float ang = pos * inv;
        float cs = __cosf(ang), sn = __sinf(ang);
        float x1 = row[QL + KL + 2 * i];
        float x2 = row[QL + KL + 2 * i + 1];
        kpe[(size_t)t * DR + 2 * i]     = (bf16)(x1 * cs - x2 * sn);
        kpe[(size_t)t * DR + 2 * i + 1] = (bf16)(x2 * cs + x1 * sn);
    }
}

// ---------------------------------------------------------------------------
// RoPE on q_pe in the bf16 q buffer [TOK, NH*192].
// ---------------------------------------------------------------------------
__global__ __launch_bounds__(256) void rope_q(bf16* __restrict__ q,
                                              const int* __restrict__ positions)
{
    const int t = blockIdx.x;
    const float pos = (float)positions[t];
    for (int p = threadIdx.x; p < NH * (DR / 2); p += 256) {
        int head = p >> 5, i = p & 31;
        float inv = __expf(-((2.0f * i) / (float)DR) * 9.210340371976184f);
        float ang = pos * inv;
        float cs = __cosf(ang), sn = __sinf(ang);
        size_t base = (size_t)t * (NH * QD) + (size_t)head * QD + DN;
        float x1 = (float)q[base + 2 * i], x2 = (float)q[base + 2 * i + 1];
        q[base + 2 * i]     = (bf16)(x1 * cs - x2 * sn);
        q[base + 2 * i + 1] = (bf16)(x2 * cs + x1 * sn);
    }
}

// ---------------------------------------------------------------------------
// Flash-style causal MLA attention on bf16 q/kv/kpe; bf16 attn output.
// Block = (head, 128-query tile), 8 waves x 16 query rows; 32-key chunks.
// K tile staged by TDM (2 descriptors: kv-slice + kpe, both padded to 416B
// rows); V tile transposed manually (TDM has no transpose).
// ---------------------------------------------------------------------------
__global__ __launch_bounds__(256) void mla_attention(
    const bf16* __restrict__ q,    // [TOK, NH*QD]
    const bf16* __restrict__ kv,   // [TOK, NH*KVD]
    const bf16* __restrict__ kpe,  // [TOK, DR]
    bf16* __restrict__ attn)       // [TOK, NH*DV]
{
    __shared__ __align__(16) bf16  Kt[32][QD + 16];  // [key][d], 416B rows
    __shared__ __align__(16) bf16  Vt[DV][48];       // [col][key] transposed
    __shared__ __align__(16) float Sw[8][16][33];
    __shared__ __align__(16) bf16  Pw[8][16][48];
    __shared__ __align__(16) float Aux[8][16];

    const int h     = blockIdx.x;
    const int qbase = blockIdx.y * 128;
    const int tid   = threadIdx.x;
    const int lane  = tid & 31;
    const int w     = tid >> 5;
    const int l16   = lane & 15;
    const int kha   = (lane < 16) ? 0 : 8;
    const int khb   = (lane < 16) ? 0 : 16;
    const int rh    = (lane < 16) ? 0 : 8;
    const int qrow  = qbase + w * 16 + l16;
    const float scale = 0.07216878364870323f;   // 1/sqrt(192)

    // Q fragments: two contiguous 16B bf16 runs each.
    v16bf qf[6];
    {
        const bf16* qp = q + (size_t)qrow * (NH * QD) + (size_t)h * QD;
#pragma unroll
        for (int c = 0; c < 6; ++c) {
            union { v16bf v; v8bf h2[2]; } u;
            u.h2[0] = *(const v8bf*)(qp + c * 32 + kha);
            u.h2[1] = *(const v8bf*)(qp + c * 32 + kha + 16);
            qf[c] = u.v;
        }
    }

    v8f o[8];
#pragma unroll
    for (int t = 0; t < 8; ++t) o[t] = v8f_zero();
    float m_r = -3.0e38f, l_r = 0.0f;

    const int kend = qbase + 128;
    for (int kc = 0; kc < kend; kc += 32) {
#if HAVE_TDM
        if (w == 0) {
            // kv slice: 128 elems (256B=64DW) per key row -> pad 40DW(160B)
            tdm_load_2d((unsigned)(uintptr_t)&Kt[0][0],
                        kv + (size_t)kc * (NH * KVD) + (size_t)h * KVD,
                        128u, 32u, (unsigned)(NH * KVD), 128u, 32u, 5u, 39u);
            // kpe slice: 64 elems (128B=32DW) per row -> pad 72DW(288B)
            tdm_load_2d((unsigned)(uintptr_t)&Kt[0][DN],
                        kpe + (size_t)kc * DR,
                        64u, 32u, (unsigned)DR, 64u, 32u, 4u, 71u);
        }
#else
        for (int idx = tid; idx < 32 * 16; idx += 256) {
            int s = idx >> 4, c8 = (idx & 15) * 8;
            *(v8bf*)&Kt[s][c8] =
                *(const v8bf*)(kv + (size_t)(kc + s) * (NH * KVD) +
                               (size_t)h * KVD + c8);
        }
        for (int idx = tid; idx < 32 * 8; idx += 256) {
            int s = idx >> 3, c8 = (idx & 7) * 8;
            *(v8bf*)&Kt[s][DN + c8] =
                *(const v8bf*)(kpe + (size_t)(kc + s) * DR + c8);
        }
#endif
        // V transposed: Vt[col][key]
        for (int idx = tid; idx < 32 * 16; idx += 256) {
            int s = idx >> 4, c8 = (idx & 15) * 8;
            v8bf v = *(const v8bf*)(kv + (size_t)(kc + s) * (NH * KVD) +
                                    (size_t)h * KVD + DN + c8);
#pragma unroll
            for (int e = 0; e < 8; ++e) Vt[c8 + e][s] = v[e];
        }
#if HAVE_TDM
        if (w == 0) __builtin_amdgcn_s_wait_tensorcnt(0);
#endif
        __syncthreads();

        // S = Q K^T for this 32-key chunk (two 16-key halves).
#pragma unroll
        for (int half = 0; half < 2; ++half) {
            v8f sacc = v8f_zero();
            const int key = half * 16 + l16;
#pragma unroll
            for (int c = 0; c < 6; ++c) {
                v16bf bf = load_frag_b(&Kt[key][c * 32 + kha]);
                sacc = __builtin_amdgcn_wmma_f32_16x16x32_bf16(
                    false, qf[c], false, bf, (short)0, sacc, false, false);
            }
#pragma unroll
            for (int i = 0; i < 8; ++i) {
                int r = rh + i;
                int col = half * 16 + l16;
                int spos = kc + col;
                int qpos = qbase + w * 16 + r;
                float sv = sacc[i] * scale;
                if (spos > qpos) sv = -3.0e38f;
                Sw[w][r][col] = sv;
            }
        }

        // Online softmax bookkeeping (one row per lane in lanes 0..15).
        if (lane < 16) {
            int r = l16;
            float mx = -3.0e38f;
            for (int c2 = 0; c2 < 32; ++c2) mx = fmaxf(mx, Sw[w][r][c2]);
            float m_new = fmaxf(m_r, mx);
            float alpha = __expf(m_r - m_new);
            float ssum = 0.0f;
            for (int c2 = 0; c2 < 32; ++c2) {
                float p = __expf(Sw[w][r][c2] - m_new);
                ssum += p;
                Pw[w][r][c2] = (bf16)p;
            }
            l_r = l_r * alpha + ssum;
            m_r = m_new;
            Aux[w][r] = alpha;
        }

        // Rescale accumulators by per-row alpha.
        {
            float4 a0 = *(const float4*)&Aux[w][rh];
            float4 a1 = *(const float4*)&Aux[w][rh + 4];
            float af[8] = {a0.x, a0.y, a0.z, a0.w, a1.x, a1.y, a1.z, a1.w};
#pragma unroll
            for (int t = 0; t < 8; ++t)
#pragma unroll
                for (int i = 0; i < 8; ++i) o[t][i] *= af[i];
        }

        // O += P @ V
        v16bf pf = load_frag_a(&Pw[w][l16][kha]);
#pragma unroll
        for (int t = 0; t < 8; ++t) {
            v16bf vf = load_frag_b(&Vt[t * 16 + l16][kha]);
            o[t] = __builtin_amdgcn_wmma_f32_16x16x32_bf16(
                false, pf, false, vf, (short)0, o[t], false, false);
        }
        __syncthreads();
    }

    if (lane < 16) Aux[w][l16] = 1.0f / l_r;
    {
        float4 a0 = *(const float4*)&Aux[w][rh];
        float4 a1 = *(const float4*)&Aux[w][rh + 4];
        float lf[8] = {a0.x, a0.y, a0.z, a0.w, a1.x, a1.y, a1.z, a1.w};
#pragma unroll
        for (int t = 0; t < 8; ++t)
#pragma unroll
            for (int i = 0; i < 8; ++i) {
                int row = qbase + w * 16 + rh + i;
                int col = t * 16 + l16;
                attn[(size_t)row * (NH * DV) + (size_t)h * DV + col] =
                    (bf16)(o[t][i] * lf[i]);
            }
    }
}

// ---------------------------------------------------------------------------
// Workspace layout (all sizes multiples of 256 bytes)
// ---------------------------------------------------------------------------
constexpr size_t OFF_QKV   = 0;                                       // f32 [T,2112]
constexpr size_t OFF_QAN   = OFF_QKV   + (size_t)TOK * QKV_N * 4;     // bf16 [T,1536]
constexpr size_t OFF_KVAN  = OFF_QAN   + (size_t)TOK * QL * 2;        // bf16 [T,512]
constexpr size_t OFF_KPE   = OFF_KVAN  + (size_t)TOK * KL * 2;        // bf16 [T,64]
constexpr size_t OFF_HB    = OFF_KPE   + (size_t)TOK * DR * 2;        // bf16 [T,5120]
constexpr size_t OFF_QB    = OFF_HB    + (size_t)TOK * HID * 2;       // bf16 [T,6144]
constexpr size_t OFF_KVB   = OFF_QB    + (size_t)TOK * NH * QD * 2;   // bf16 [T,8192]
constexpr size_t OFF_ATTNB = OFF_KVB   + (size_t)TOK * NH * KVD * 2;  // bf16 [T,4096]
constexpr size_t OFF_WT1   = OFF_ATTNB + (size_t)TOK * NH * DV * 2;   // bf16 [2112,5120]
constexpr size_t OFF_WTQ   = OFF_WT1   + (size_t)QKV_N * HID * 2;     // bf16 [6144,1536]
constexpr size_t OFF_WTKV  = OFF_WTQ   + (size_t)(NH * QD) * QL * 2;  // bf16 [8192,512]
constexpr size_t OFF_WTO   = OFF_WTKV  + (size_t)(NH * KVD) * KL * 2; // bf16 [5120,4096]

extern "C" void kernel_launch(void* const* d_in, const int* in_sizes, int n_in,
                              void* d_out, int out_size, void* d_ws, size_t ws_size,
                              hipStream_t stream) {
    const int*   positions = (const int*)d_in[0];
    const float* hidden    = (const float*)d_in[1];
    const float* w_qkv_a   = (const float*)d_in[2];
    const float* q_a_ln_w  = (const float*)d_in[3];
    const float* w_q_b     = (const float*)d_in[4];
    const float* kv_a_ln_w = (const float*)d_in[5];
    const float* w_kv_b    = (const float*)d_in[6];
    const float* w_o       = (const float*)d_in[7];
    float* out = (float*)d_out;

    char* ws = (char*)d_ws;
    float* qkv   = (float*)(ws + OFF_QKV);
    bf16*  qan   = (bf16*) (ws + OFF_QAN);
    bf16*  kvan  = (bf16*) (ws + OFF_KVAN);
    bf16*  kpe   = (bf16*) (ws + OFF_KPE);
    bf16*  hb    = (bf16*) (ws + OFF_HB);
    bf16*  qbuf  = (bf16*) (ws + OFF_QB);
    bf16*  kvbuf = (bf16*) (ws + OFF_KVB);
    bf16*  attnb = (bf16*) (ws + OFF_ATTNB);
    bf16*  wt1   = (bf16*) (ws + OFF_WT1);
    bf16*  wtq   = (bf16*) (ws + OFF_WTQ);
    bf16*  wtkv  = (bf16*) (ws + OFF_WTKV);
    bf16*  wto   = (bf16*) (ws + OFF_WTO);

    // 0) One-time precision/layout prep: weights -> bf16 [N,K]; hidden -> bf16
    transpose_cvt<<<dim3(QKV_N / 32, HID / 32), 256, 0, stream>>>(w_qkv_a, wt1, HID, QKV_N);
    transpose_cvt<<<dim3((NH * QD) / 32, QL / 32), 256, 0, stream>>>(w_q_b, wtq, QL, NH * QD);
    transpose_cvt<<<dim3((NH * KVD) / 32, KL / 32), 256, 0, stream>>>(w_kv_b, wtkv, KL, NH * KVD);
    transpose_cvt<<<dim3(HID / 32, (NH * DV) / 32), 256, 0, stream>>>(w_o, wto, NH * DV, HID);
    cvt_f32_bf16<<<2048, 256, 0, stream>>>(hidden, hb, TOK * HID / 4);

    // 1) qkv = hidden @ w_qkv_a
    gemm_bf16<float><<<dim3((QKV_N + 127) / 128, TOK / 128), 256, 0, stream>>>(
        hb, wt1, qkv, TOK, QKV_N, HID);

    // 2) RMSNorm q_a/kv_a -> bf16, RoPE k_pe -> bf16
    prep_norm_rope<<<TOK, 256, 0, stream>>>(qkv, q_a_ln_w, kv_a_ln_w, positions,
                                            qan, kvan, kpe);

    // 3) q = q_a_n @ w_q_b  (bf16 out)
    gemm_bf16<bf16><<<dim3((NH * QD) / 128, TOK / 128), 256, 0, stream>>>(
        qan, wtq, qbuf, TOK, NH * QD, QL);

    // 4) kv = kv_a_n @ w_kv_b  (bf16 out)
    gemm_bf16<bf16><<<dim3((NH * KVD) / 128, TOK / 128), 256, 0, stream>>>(
        kvan, wtkv, kvbuf, TOK, NH * KVD, KL);

    // 5) RoPE on q_pe (in-place, bf16)
    rope_q<<<TOK, 256, 0, stream>>>(qbuf, positions);

    // 6) Causal flash attention -> bf16 attn
    mla_attention<<<dim3(NH, TOK / 128), 256, 0, stream>>>(qbuf, kvbuf, kpe, attnb);

    // 7) out = attn @ w_o  (f32 out)
    gemm_bf16<float><<<dim3(HID / 128, TOK / 128), 256, 0, stream>>>(
        attnb, wto, out, TOK, HID, NH * DV);
}
